// GAT_33938831573044
// MI455X (gfx1250) — compile-verified, matrix operand order checked by reading, not run on hardware
//
#include <hip/hip_runtime.h>

typedef __attribute__((ext_vector_type(16))) __bf16        v16bf;
typedef __attribute__((ext_vector_type(8)))  float         v8f;
typedef __attribute__((ext_vector_type(8)))  unsigned int  v8u;

#define NEG_SLOPE 0.2f

// ---------------- helpers ----------------
__device__ __forceinline__ unsigned short f2bf(float f) {
  unsigned u = __float_as_uint(f);
  u += 0x7fffu + ((u >> 16) & 1u);          // round-to-nearest-even
  return (unsigned short)(u >> 16);
}
// order-preserving float<->uint mapping (for atomic float max via atomicMax(u32))
__device__ __forceinline__ unsigned f2ord(float f) {
  unsigned b = __float_as_uint(f);
  return (b & 0x80000000u) ? ~b : (b | 0x80000000u);
}
__device__ __forceinline__ float ord2f(unsigned u) {
  unsigned b = (u & 0x80000000u) ? (u ^ 0x80000000u) : ~u;
  return __uint_as_float(b);
}

// ---------------- elementwise / prep kernels ----------------
// pad+convert fp32 [srcR,srcC] -> bf16 [dstR,dstC] (zero pad), optional ReLU
__global__ __launch_bounds__(256)
void pad_convert_k(const float* __restrict__ src, unsigned short* __restrict__ dst,
                   int srcR, int srcC, int dstR, int dstC, int relu) {
  long long t = (long long)blockIdx.x * blockDim.x + threadIdx.x;
  long long tot = (long long)dstR * dstC;
  if (t >= tot) return;
  int r = (int)(t / dstC), c = (int)(t % dstC);
  float v = 0.f;
  if (r < srcR && c < srcC) v = src[(size_t)r * srcC + c];
  if (relu) v = v > 0.f ? v : 0.f;
  dst[t] = f2bf(v);
}

__global__ __launch_bounds__(256)
void fill_u32_k(unsigned* __restrict__ p, unsigned val, long long n) {
  long long t = (long long)blockIdx.x * blockDim.x + threadIdx.x;
  if (t < n) p[t] = val;
}

__global__ __launch_bounds__(256)
void init_emax_k(unsigned* __restrict__ p, long long n) {
  long long t = (long long)blockIdx.x * blockDim.x + threadIdx.x;
  if (t < n) p[t] = f2ord(-3.0e38f);
}

// out[n, j] = bias[j]  (aggregation accumulates on top)
__global__ __launch_bounds__(256)
void init_bias_k(float* __restrict__ out, const float* __restrict__ b, int N, int M) {
  long long t = (long long)blockIdx.x * blockDim.x + threadIdx.x;
  if (t >= (long long)N * M) return;
  out[t] = b[(int)(t % M)];
}

// ---------------- WMMA GEMM: C[N,M] = A_bf16[N,Kp] @ W_bf16[Kp,M], fp32 accum ----
// One wave owns one 16-wide column panel. B fragments for the *entire* K range
// live in registers (<=56 VGPRs) and are reused across RT=5 row tiles. Both KS
// (= Kp/32) and M are compile-time constants: all B-load / C-store offsets are
// 24-bit instruction immediates (no runtime address chains), and per row tile
// all KS A-fragments are loaded as one clause before the KS back-to-back
// v_wmma_f32_16x16x32_bf16 ops. N rows must be a multiple of 80 (50000=80*625).
template <int KS, int M>
__global__ __launch_bounds__(128)
void wmma_gemm_k(const unsigned short* __restrict__ A,
                 const unsigned short* __restrict__ W,
                 float* __restrict__ C) {
  constexpr int Kp = KS * 32;
  constexpr int RT = 5;
  const int lane = threadIdx.x & 31;
  const int wave = threadIdx.x >> 5;
  const int nTile = blockIdx.y * 4 + wave;
  if (nTile * 16 >= M) return;              // wave-uniform guard (EXEC all-1 inside)
  const int nBase = nTile * 16;
  const int ml = lane & 15;                 // col (B/C/D) or row (A) within tile
  const int kh = lane >> 4;                 // K-half selector

  // ---- load B fragments once for all K steps (immediate offsets off wp) ----
  v8u breg[KS];
  const unsigned short* wp = W + (size_t)(kh << 3) * M + nBase + ml;
#pragma unroll
  for (int ks = 0; ks < KS; ++ks) {
#pragma unroll
    for (int v = 0; v < 8; ++v) {
      const int k = ks * 32 + ((v >> 2) << 4) + ((v & 3) << 1);  // kh folded in wp
      unsigned lo = wp[k * M];
      unsigned hi = wp[(k + 1) * M];
      breg[ks][v] = lo | (hi << 16);
    }
  }

  const int rowBase = blockIdx.x * (16 * RT);
#pragma unroll
  for (int rt = 0; rt < RT; ++rt) {
    const int rowTile = rowBase + rt * 16;
    const unsigned short* ap = A + (size_t)(rowTile + ml) * Kp + (kh << 3);
    if (rt + 1 < RT)                         // warm L0/L2 for the next row tile
      __builtin_prefetch(ap + 16 * Kp, 0, 1);

    // ---- load ALL A fragments for this row tile first (one load clause) ----
    v8u areg[KS];
#pragma unroll
    for (int ks = 0; ks < KS; ++ks)
#pragma unroll
      for (int v = 0; v < 8; ++v)
        areg[ks][v] = *(const unsigned int*)(ap + ks * 32 + ((v >> 2) << 4) + ((v & 3) << 1));

    // ---- KS chained WMMAs ----
    v8f acc = {};
#pragma unroll
    for (int ks = 0; ks < KS; ++ks)
      acc = __builtin_amdgcn_wmma_f32_16x16x32_bf16(
          /*neg_a=*/false, __builtin_bit_cast(v16bf, areg[ks]),
          /*neg_b=*/false, __builtin_bit_cast(v16bf, breg[ks]),
          /*c_mod=*/(short)0, acc, /*reuse_a=*/false, /*reuse_b=*/false);

    // ---- store: immediate row-stride offsets off one base ----
    float* cp = C + (size_t)(rowTile + (kh << 3)) * M + nBase + ml;
#pragma unroll
    for (int r = 0; r < 8; ++r)
      cp[r * M] = acc[r];
  }
}

// ---------------- attention kernels ----------------
// alpha_s[n,h] = sum_c h[n,h,c]*a_src[h,c]; alpha_d likewise
__global__ __launch_bounds__(256)
void alpha_k(const float* __restrict__ h, const float* __restrict__ a_src,
             const float* __restrict__ a_dst, float* __restrict__ as,
             float* __restrict__ ad, int N, int H, int C) {
  int t = blockIdx.x * blockDim.x + threadIdx.x;
  if (t >= N * H) return;
  int n = t / H, hd = t % H;
  const float* hp  = h + (size_t)n * H * C + (size_t)hd * C;
  const float* asp = a_src + hd * C;
  const float* adp = a_dst + hd * C;
  float s = 0.f, d = 0.f;
  for (int c = 0; c < C; ++c) { float v = hp[c]; s += v * asp[c]; d += v * adp[c]; }
  as[t] = s; ad[t] = d;
}

__device__ __forceinline__ void edge_sd(const int* ei, int E, long long e, int& s, int& d) {
  if (e < E) { s = ei[e]; d = ei[E + e]; }
  else       { s = (int)(e - E); d = s; }           // self loops
}

// pass 1: segment max over dst
__global__ __launch_bounds__(256)
void edge_max_k(const int* __restrict__ ei, int E, int N, int H,
                const float* __restrict__ as, const float* __restrict__ ad,
                unsigned* __restrict__ emax) {
  long long t = (long long)blockIdx.x * blockDim.x + threadIdx.x;
  long long tot = (long long)(E + N) * H;
  if (t >= tot) return;
  long long e = t / H; int hd = (int)(t % H);
  int s, d; edge_sd(ei, E, e, s, d);
  float v = as[(size_t)s * H + hd] + ad[(size_t)d * H + hd];
  v = v > 0.f ? v : NEG_SLOPE * v;
  atomicMax(&emax[(size_t)d * H + hd], f2ord(v));
}

// pass 2: denom[dst,h] += exp(e - emax[dst,h])
__global__ __launch_bounds__(256)
void edge_sum_k(const int* __restrict__ ei, int E, int N, int H,
                const float* __restrict__ as, const float* __restrict__ ad,
                const unsigned* __restrict__ emax, float* __restrict__ den) {
  long long t = (long long)blockIdx.x * blockDim.x + threadIdx.x;
  long long tot = (long long)(E + N) * H;
  if (t >= tot) return;
  long long e = t / H; int hd = (int)(t % H);
  int s, d; edge_sd(ei, E, e, s, d);
  float v = as[(size_t)s * H + hd] + ad[(size_t)d * H + hd];
  v = v > 0.f ? v : NEG_SLOPE * v;
  float ee = expf(v - ord2f(emax[(size_t)d * H + hd]));
  atomicAdd(&den[(size_t)d * H + hd], ee);
}

// pass 3: out[dst,h,:] += alpha * h[src,h,:]
__global__ __launch_bounds__(256)
void edge_agg_k(const int* __restrict__ ei, int E, int N, int H, int C,
                const float* __restrict__ as, const float* __restrict__ ad,
                const unsigned* __restrict__ emax, const float* __restrict__ den,
                const float* __restrict__ h, float* __restrict__ out) {
  long long t = (long long)blockIdx.x * blockDim.x + threadIdx.x;
  long long tot = (long long)(E + N) * H;
  if (t >= tot) return;
  long long e = t / H; int hd = (int)(t % H);
  int s, d; edge_sd(ei, E, e, s, d);
  float v = as[(size_t)s * H + hd] + ad[(size_t)d * H + hd];
  v = v > 0.f ? v : NEG_SLOPE * v;
  float ee = expf(v - ord2f(emax[(size_t)d * H + hd]));
  float alpha = ee / (den[(size_t)d * H + hd] + 1e-16f);
  const float* hp = h + (size_t)s * H * C + (size_t)hd * C;
  float* op = out + (size_t)d * H * C + (size_t)hd * C;
  for (int c = 0; c < C; ++c) atomicAdd(&op[c], alpha * hp[c]);
}

// ---------------- host-side layer driver ----------------
static inline unsigned cdiv_ll(long long a, int b) { return (unsigned)((a + b - 1) / b); }

template <int KS, int M>
static void gat_layer(const unsigned short* featBf,
                      const float* W, int K,
                      const float* a_src, const float* a_dst, const float* bias,
                      const int* ei, int E, int N, int H, int C,
                      unsigned short* wBf, float* hbuf, float* out,
                      float* as, float* ad, unsigned* emax, float* den,
                      hipStream_t st) {
  constexpr int Kp = KS * 32;
  // weights -> bf16 padded [Kp, M]
  pad_convert_k<<<cdiv_ll((long long)Kp * M, 256), 256, 0, st>>>(W, wBf, K, M, Kp, M, 0);
  // GEMM: h = feat @ W   (RT=5 row tiles per wave -> grid.x = N/80)
  dim3 gg((unsigned)(N / 80), (unsigned)((M / 16 + 3) / 4));
  wmma_gemm_k<KS, M><<<gg, 128, 0, st>>>(featBf, wBf, hbuf);
  // attention halves
  alpha_k<<<cdiv_ll((long long)N * H, 256), 256, 0, st>>>(hbuf, a_src, a_dst, as, ad, N, H, C);
  // segment softmax + aggregation
  long long nh = (long long)N * H;
  long long eh = (long long)(E + N) * H;
  init_emax_k<<<cdiv_ll(nh, 256), 256, 0, st>>>(emax, nh);
  edge_max_k<<<cdiv_ll(eh, 256), 256, 0, st>>>(ei, E, N, H, as, ad, emax);
  fill_u32_k<<<cdiv_ll(nh, 256), 256, 0, st>>>((unsigned*)den, 0u, nh);
  edge_sum_k<<<cdiv_ll(eh, 256), 256, 0, st>>>(ei, E, N, H, as, ad, emax, den);
  init_bias_k<<<cdiv_ll((long long)N * M, 256), 256, 0, st>>>(out, bias, N, M);
  edge_agg_k<<<cdiv_ll(eh, 256), 256, 0, st>>>(ei, E, N, H, C, as, ad, emax, den, hbuf, out);
}

extern "C" void kernel_launch(void* const* d_in, const int* in_sizes, int n_in,
                              void* d_out, int out_size, void* d_ws, size_t ws_size,
                              hipStream_t stream) {
  const float* x   = (const float*)d_in[0];
  const float* W1  = (const float*)d_in[1];
  const float* a1s = (const float*)d_in[2];
  const float* a1d = (const float*)d_in[3];
  const float* b1  = (const float*)d_in[4];
  const float* W2  = (const float*)d_in[5];
  const float* a2s = (const float*)d_in[6];
  const float* a2d = (const float*)d_in[7];
  const float* b2  = (const float*)d_in[8];
  const float* W3  = (const float*)d_in[9];
  const float* a3s = (const float*)d_in[10];
  const float* a3d = (const float*)d_in[11];
  const float* b3  = (const float*)d_in[12];
  const int*   ei  = (const int*)d_in[13];   // int32 (JAX x64 disabled)

  const int F_IN = 129, HID = 32, NCLS = 40;
  const int N = in_sizes[0] / F_IN;          // 50000 (= 80*625)
  const int E = in_sizes[13] / 2;            // 500000

  // workspace layout (bytes)
  char* ws = (char*)d_ws;
  unsigned short* FEAT = (unsigned short*)(ws + 0);          // N*256 bf16 = 25.6 MB
  unsigned short* WBF  = (unsigned short*)(ws + 25600000);   // <=1 MB weights bf16
  float*    HB   = (float*)(ws + 26648576);                  // N*256 f32 = 51.2 MB
  float*    OUT  = (float*)(ws + 77848576);                  // N*256 f32 = 51.2 MB
  float*    AS   = (float*)(ws + 129048576);                 // N*8 f32
  float*    AD   = (float*)(ws + 130648576);
  unsigned* EMAX = (unsigned*)(ws + 132248576);
  float*    DEN  = (float*)(ws + 133848576);

  // ---- layer 1: [N,129] -> [N,7*32]  (Kp=160, M=224) ----
  pad_convert_k<<<cdiv_ll((long long)N * 160, 256), 256, 0, stream>>>(x, FEAT, N, F_IN, N, 160, 0);
  gat_layer<5, 224>(FEAT, W1, F_IN, a1s, a1d, b1, ei, E, N, 7, HID,
                    WBF, HB, OUT, AS, AD, EMAX, DEN, stream);

  // ---- layer 2: relu -> [N,224] -> [N,6*32]  (Kp=224, M=192) ----
  pad_convert_k<<<cdiv_ll((long long)N * 224, 256), 256, 0, stream>>>(OUT, FEAT, N, 224, N, 224, 1);
  gat_layer<7, 192>(FEAT, W2, 224, a2s, a2d, b2, ei, E, N, 6, HID,
                    WBF, HB, OUT, AS, AD, EMAX, DEN, stream);

  // ---- layer 3: relu -> [N,192] -> [N,6*40] directly into d_out  (Kp=192, M=240) ----
  pad_convert_k<<<cdiv_ll((long long)N * 192, 256), 256, 0, stream>>>(OUT, FEAT, N, 192, N, 192, 1);
  gat_layer<6, 240>(FEAT, W3, 192, a3s, a3d, b3, ei, E, N, 6, NCLS,
                    WBF, HB, (float*)d_out, AS, AD, EMAX, DEN, stream);
}
